// SentenceDecoder_30614526886209
// MI455X (gfx1250) — matrix operand order repaired and checked by reading, not compiled, not run on hardware
//
#include <hip/hip_runtime.h>
#include <hip/hip_bf16.h>
#include <math.h>

#define B_    64
#define S_    196
#define H_    512
#define E_    512
#define VP_   8
#define LAB_  16
#define V_    10000
#define L_    32
#define T_    31
#define IN_   1560          // H + VP + LAB + E + E
#define KCAT  2072          // IN_ + H_
#define KCATP 2080          // padded to multiple of 32
#define G4H   2048          // 4*H

typedef __attribute__((ext_vector_type(16))) __bf16 v16bf;
typedef __attribute__((ext_vector_type(8)))  __bf16 v8bf;
typedef __attribute__((ext_vector_type(8)))  float  v8f;

union V16U { v16bf v; struct { v8bf lo, hi; } p; };

// ---------------------------------------------------------------------------
// WMMA GEMM:  C[M x N] (f32) = act( A[M x Kp] (bf16, row-major, stride lda,
// column offset aoff) * Bt[N x Kp]^T (bf16, pre-transposed: row n holds K) + bias )
// Wave tile = 64 rows x 16 cols (4 A-frags share one B-frag). Requires M % 64 == 0.
// ---------------------------------------------------------------------------
__global__ void gemm_bf16_tn(const __bf16* __restrict__ A, const __bf16* __restrict__ Bt,
                             const float* __restrict__ bias, float* __restrict__ C,
                             __bf16* __restrict__ Cbf,
                             int M, int N, int Kp, int lda, int aoff, int ldc, int act)
{
  const int wave = (int)((blockIdx.x * blockDim.x + threadIdx.x) >> 5);
  const int lane = (int)(threadIdx.x & 31);
  const int mtiles = M >> 6;
  const int ntiles = (N + 15) >> 4;
  if (wave >= mtiles * ntiles) return;
  const int mi   = wave % mtiles;
  const int ni   = wave / mtiles;
  const int row0 = mi << 6;
  const int half = lane >> 4;      // 0: lanes 0-15, 1: lanes 16-31
  const int mr   = lane & 15;
  const int ncol = (ni << 4) + mr;

  const __bf16* a0p = A + (size_t)(row0 + mr) * lda + aoff;
  const __bf16* a1p = a0p + (size_t)16 * lda;
  const __bf16* a2p = a0p + (size_t)32 * lda;
  const __bf16* a3p = a0p + (size_t)48 * lda;
  const __bf16* bp  = Bt + (size_t)(ncol < N ? ncol : N - 1) * Kp;

  const int kbA = half << 3;   // A: K sub-offset 0 or 8 (pairs in VGPR0-3 / +16 in VGPR4-7)
  const int kbB = half << 3;   // B: same K order as A (8h+0..7, then 16+8h+0..7)

  v8f acc0 = {}, acc1 = {}, acc2 = {}, acc3 = {};

  for (int k0 = 0; k0 < Kp; k0 += 32) {
    V16U b;
    b.p.lo = *(const v8bf*)(bp + k0 + kbB);
    b.p.hi = *(const v8bf*)(bp + k0 + kbB + 16);
    const int ka = k0 + kbA;
    V16U a0, a1, a2, a3;
    a0.p.lo = *(const v8bf*)(a0p + ka);  a0.p.hi = *(const v8bf*)(a0p + ka + 16);
    a1.p.lo = *(const v8bf*)(a1p + ka);  a1.p.hi = *(const v8bf*)(a1p + ka + 16);
    a2.p.lo = *(const v8bf*)(a2p + ka);  a2.p.hi = *(const v8bf*)(a2p + ka + 16);
    a3.p.lo = *(const v8bf*)(a3p + ka);  a3.p.hi = *(const v8bf*)(a3p + ka + 16);
    acc0 = __builtin_amdgcn_wmma_f32_16x16x32_bf16(false, a0.v, false, b.v, (short)0, acc0, false, false);
    acc1 = __builtin_amdgcn_wmma_f32_16x16x32_bf16(false, a1.v, false, b.v, (short)0, acc1, false, false);
    acc2 = __builtin_amdgcn_wmma_f32_16x16x32_bf16(false, a2.v, false, b.v, (short)0, acc2, false, false);
    acc3 = __builtin_amdgcn_wmma_f32_16x16x32_bf16(false, a3.v, false, b.v, (short)0, acc3, false, false);
  }

  if (ncol >= N) return;
  const float bv = bias ? bias[ncol] : 0.f;
  v8f accs[4] = {acc0, acc1, acc2, acc3};
  for (int mt = 0; mt < 4; ++mt) {
    for (int v = 0; v < 8; ++v) {
      const int r = row0 + (mt << 4) + v + (half << 3);   // C/D: VGPR v -> row v (+8 for upper half-wave)
      float x = accs[mt][v] + bv;
      if (act == 1) x = tanhf(x);
      C[(size_t)r * ldc + ncol] = x;
      if (Cbf) Cbf[(size_t)r * ldc + ncol] = (__bf16)x;
    }
  }
}

// ------------------------- setup / conversion kernels -----------------------

__global__ void mean_kernel(const float* __restrict__ image, float* __restrict__ img_mean,
                            __bf16* __restrict__ img_mean_bf)
{
  const int b = blockIdx.x;
  for (int h = threadIdx.x; h < H_; h += blockDim.x) {
    const float* p = image + (size_t)b * S_ * H_ + h;
    float s = 0.f;
    for (int i = 0; i < S_; ++i) s += p[(size_t)i * H_];
    const float m = s * (1.f / (float)S_);
    img_mean[b * H_ + h] = m;
    img_mean_bf[b * H_ + h] = (__bf16)m;
  }
}

// fill static part of x_full: [img_mean(512) | vp(8) | label(16) | topic(512)] + zero tail pad
__global__ void static_init_kernel(__bf16* __restrict__ x_full, const float* __restrict__ img_mean,
                                   const float* __restrict__ vp, const float* __restrict__ label,
                                   const float* __restrict__ topic)
{
  const int idx = blockIdx.x * blockDim.x + threadIdx.x;
  if (idx >= B_ * 1056) return;
  const int b = idx / 1056, j = idx % 1056;
  __bf16* xr = x_full + (size_t)b * KCATP;
  if (j < 512)        xr[j] = (__bf16)img_mean[b * H_ + j];
  else if (j < 520)   xr[j] = (__bf16)vp[b * VP_ + (j - 512)];
  else if (j < 536)   xr[j] = (__bf16)label[b * LAB_ + (j - 520)];
  else if (j < 1048)  xr[j] = (__bf16)topic[b * E_ + (j - 536)];
  else                xr[KCAT + (j - 1048)] = (__bf16)0.f;   // pad cols 2072..2079
}

// transpose-convert f32 W[K x N] -> bf16 Wt[N x Kp] with zero K-padding
__global__ void wconv_kernel(const float* __restrict__ W, __bf16* __restrict__ Wt,
                             int K, int N, int Kp)
{
  const int idx = blockIdx.x * blockDim.x + threadIdx.x;
  if (idx >= N * Kp) return;
  const int n = idx / Kp, k = idx % Kp;
  Wt[idx] = (k < K) ? (__bf16)W[(size_t)k * N + n] : (__bf16)0.f;
}

// concat+transpose: Wcat_t[2048 x 2080] from W_ih(1560x2048) and W_hh(512x2048)
__global__ void wcat_kernel(const float* __restrict__ Wih, const float* __restrict__ Whh,
                            __bf16* __restrict__ Wt)
{
  const int idx = blockIdx.x * blockDim.x + threadIdx.x;
  if (idx >= G4H * KCATP) return;
  const int n = idx / KCATP, k = idx % KCATP;
  float v = 0.f;
  if (k < IN_)        v = Wih[(size_t)k * G4H + n];
  else if (k < KCAT)  v = Whh[(size_t)(k - IN_) * G4H + n];
  Wt[idx] = (__bf16)v;
}

__global__ void biascat_kernel(const float* a, const float* b, float* c)
{
  const int i = blockIdx.x * blockDim.x + threadIdx.x;
  if (i < G4H) c[i] = a[i] + b[i];
}

__global__ void f2bf_kernel(const float* __restrict__ in, __bf16* __restrict__ out, int n)
{
  const int i = blockIdx.x * blockDim.x + threadIdx.x;
  if (i < n) out[i] = (__bf16)in[i];
}

// ------------------------------ per-step kernels ----------------------------

// update x_full cols [1048,1560) = emb_W[text[b,t]] and [1560,2072) = h_bf
__global__ void x_update_kernel(__bf16* __restrict__ x_full, const float* __restrict__ emb_W,
                                const int* __restrict__ text, const __bf16* __restrict__ h_bf, int t)
{
  const int idx = blockIdx.x * blockDim.x + threadIdx.x;
  if (idx >= B_ * 1024) return;
  const int b = idx >> 10, j = idx & 1023;
  __bf16* xr = x_full + (size_t)b * KCATP;
  if (j < 512) {
    const int tok = text[b * L_ + t];
    xr[1048 + j] = (__bf16)emb_W[(size_t)tok * E_ + j];
  } else {
    xr[1560 + (j - 512)] = h_bf[b * H_ + (j - 512)];
  }
}

__device__ __forceinline__ float sigmoidf_(float x) { return 1.f / (1.f + expf(-x)); }

__global__ void lstm_kernel(const float* __restrict__ gates, float* __restrict__ h,
                            float* __restrict__ m, __bf16* __restrict__ h_bf)
{
  const int idx = blockIdx.x * blockDim.x + threadIdx.x;
  if (idx >= B_ * H_) return;
  const int b = idx / H_, j = idx % H_;
  const float* g = gates + (size_t)b * G4H;
  const float i_ = sigmoidf_(g[j]);
  const float f_ = sigmoidf_(g[H_ + j]);
  const float g_ = tanhf(g[2 * H_ + j]);
  const float o_ = sigmoidf_(g[3 * H_ + j]);
  const float m2 = f_ * m[idx] + i_ * g_;
  const float h2 = o_ * tanhf(m2);
  m[idx] = m2;
  h[idx] = h2;
  h_bf[idx] = (__bf16)h2;
}

__device__ __forceinline__ float warp_sum(float v) {
  for (int o = 16; o; o >>= 1) v += __shfl_down(v, o, 32);
  return v;
}
__device__ __forceinline__ float warp_max(float v) {
  for (int o = 16; o; o >>= 1) v = fmaxf(v, __shfl_down(v, o, 32));
  return v;
}
__device__ float block_red(float v, int is_max) {
  __shared__ float sh[8];
  const int wid = threadIdx.x >> 5, lane = threadIdx.x & 31;
  v = is_max ? warp_max(v) : warp_sum(v);
  if (lane == 0) sh[wid] = v;
  __syncthreads();
  if (wid == 0) {
    float x = (lane < (int)(blockDim.x >> 5)) ? sh[lane] : (is_max ? -3.4e38f : 0.f);
    x = is_max ? warp_max(x) : warp_sum(x);
    if (lane == 0) sh[0] = x;
  }
  __syncthreads();
  const float r = sh[0];
  __syncthreads();
  return r;
}

// zz[b,s] = sum_h tanh(v_proj[b,s,h] + _h[b,h]) * wz[h]  (s<196), tanh(_s+_h) row for s==196
__global__ void zz_kernel(const float* __restrict__ v_proj, const float* __restrict__ _h,
                          const float* __restrict__ _s, const float* __restrict__ wz,
                          const float* __restrict__ zb, float* __restrict__ zz)
{
  const int b = blockIdx.x / 25;
  const int wave = threadIdx.x >> 5, lane = threadIdx.x & 31;
  const int s = (blockIdx.x % 25) * 8 + wave;
  if (s > S_) return;
  const float* hb = _h + (size_t)b * H_;
  float sum = 0.f;
  if (s < S_) {
    const float* vp = v_proj + ((size_t)b * S_ + s) * H_;
    for (int h = lane; h < H_; h += 32) sum += tanhf(vp[h] + hb[h]) * wz[h];
  } else {
    const float* sb = _s + (size_t)b * H_;
    for (int h = lane; h < H_; h += 32) sum += tanhf(sb[h] + hb[h]) * wz[h];
  }
  sum = warp_sum(sum);
  if (lane == 0) zz[b * (S_ + 1) + s] = sum + zb[0];
}

// a = softmax(zz[b, :197]); also store masked attention to output
__global__ void softmax_att_kernel(const float* __restrict__ zz, float* __restrict__ a,
                                   float* __restrict__ att_out, const int* __restrict__ length, int t)
{
  const int b = blockIdx.x;
  const float* z = zz + (size_t)b * (S_ + 1);
  float mx = -3.4e38f;
  for (int s = threadIdx.x; s <= S_; s += blockDim.x) mx = fmaxf(mx, z[s]);
  mx = block_red(mx, 1);
  float sum = 0.f;
  for (int s = threadIdx.x; s <= S_; s += blockDim.x) sum += expf(z[s] - mx);
  sum = block_red(sum, 0);
  const float inv = 1.f / sum;
  const float msk = (t < length[b] - 1) ? 1.f : 0.f;
  float* ao = att_out + ((size_t)b * T_ + t) * (S_ + 1);
  for (int s = threadIdx.x; s <= S_; s += blockDim.x) {
    const float av = expf(z[s] - mx) * inv;
    a[b * (S_ + 1) + s] = av;
    ao[s] = av * msk;
  }
}

// c[b,h] = sum_s a[b,s]*image[b,s,h] + a[b,196]*hs[b,512+h];  cp_bf = bf16(c + h2)
__global__ void ctx_kernel(const float* __restrict__ a, const float* __restrict__ image,
                           const float* __restrict__ hs, const float* __restrict__ h2,
                           __bf16* __restrict__ cp_bf)
{
  const int b = blockIdx.x;
  const float* ab = a + (size_t)b * (S_ + 1);
  for (int h = threadIdx.x; h < H_; h += blockDim.x) {
    const float* im = image + (size_t)b * S_ * H_ + h;
    float acc = 0.f;
    for (int s = 0; s < S_; ++s) acc += ab[s] * im[(size_t)s * H_];
    acc += ab[S_] * hs[(size_t)b * 1024 + 512 + h];
    cp_bf[b * H_ + h] = (__bf16)(acc + h2[b * H_ + h]);
  }
}

// logp[b,t,:] = (log_softmax(logits[b,:]) / temp[b]) * mask
__global__ void logsoftmax_kernel(const float* __restrict__ logits, float* __restrict__ out,
                                  const float* __restrict__ temp, const int* __restrict__ length, int t)
{
  const int b = blockIdx.x;
  const float* lg = logits + (size_t)b * V_;
  float mx = -3.4e38f;
  for (int v = threadIdx.x; v < V_; v += blockDim.x) mx = fmaxf(mx, lg[v]);
  mx = block_red(mx, 1);
  float se = 0.f;
  for (int v = threadIdx.x; v < V_; v += blockDim.x) se += expf(lg[v] - mx);
  se = block_red(se, 0);
  const float lse = mx + logf(se);
  const float msk = (t < length[b] - 1) ? 1.f : 0.f;
  const float scale = msk / temp[b];
  float* o = out + ((size_t)b * T_ + t) * V_;
  for (int v = threadIdx.x; v < V_; v += blockDim.x) o[v] = (lg[v] - lse) * scale;
}

// ------------------------------- host side ----------------------------------

static inline void gemm_launch(hipStream_t st, const __bf16* A, const __bf16* Bt, const float* bias,
                               float* C, __bf16* Cbf, int M, int N, int Kp, int lda, int aoff,
                               int ldc, int act)
{
  const int waves = (M >> 6) * ((N + 15) >> 4);
  const int blocks = (waves + 7) >> 3;          // 8 waves (256 threads) per block
  gemm_bf16_tn<<<blocks, 256, 0, st>>>(A, Bt, bias, C, Cbf, M, N, Kp, lda, aoff, ldc, act);
}

extern "C" void kernel_launch(void* const* d_in, const int* in_sizes, int n_in,
                              void* d_out, int out_size, void* d_ws, size_t ws_size,
                              hipStream_t stream)
{
  (void)in_sizes; (void)n_in; (void)out_size; (void)ws_size;

  const float* image    = (const float*)d_in[0];
  const float* view_pos = (const float*)d_in[1];
  const float* label    = (const float*)d_in[2];
  const float* topic    = (const float*)d_in[3];
  const float* temp     = (const float*)d_in[4];
  const float* emb_W    = (const float*)d_in[5];
  const float* fc_h_W   = (const float*)d_in[6];
  const float* fc_h_b   = (const float*)d_in[7];
  const float* fc_m_W   = (const float*)d_in[8];
  const float* fc_m_b   = (const float*)d_in[9];
  const float* W_ih     = (const float*)d_in[10];
  const float* W_hh     = (const float*)d_in[11];
  const float* b_ih     = (const float*)d_in[12];
  const float* b_hh     = (const float*)d_in[13];
  const float* att_fc_W = (const float*)d_in[14];
  const float* att_fc_b = (const float*)d_in[15];
  const float* att_v_W  = (const float*)d_in[16];
  const float* att_v_b  = (const float*)d_in[17];
  const float* att_hh_W = (const float*)d_in[18];
  const float* att_hh_b = (const float*)d_in[19];
  const float* att_s_W  = (const float*)d_in[20];
  const float* att_s_b  = (const float*)d_in[21];
  const float* att_z_W  = (const float*)d_in[22];
  const float* att_z_b  = (const float*)d_in[23];
  const float* fc_p_W   = (const float*)d_in[24];
  const float* fc_p_b   = (const float*)d_in[25];
  const int*   text     = (const int*)d_in[26];
  const int*   length   = (const int*)d_in[27];

  float* att_out  = (float*)d_out;
  float* logp_out = att_out + (size_t)B_ * T_ * (S_ + 1);

  // bump allocator over workspace (256B aligned)
  char* wp = (char*)d_ws;
  auto alloc = [&](size_t bytes) -> void* {
    void* r = (void*)wp;
    wp += (bytes + 255) & ~(size_t)255;
    return r;
  };
  __bf16* x_full     = (__bf16*)alloc((size_t)B_ * KCATP * 2);
  __bf16* Wcat_t     = (__bf16*)alloc((size_t)G4H * KCATP * 2);
  float*  bias_cat   = (float*) alloc((size_t)G4H * 4);
  __bf16* fc_h_Wt    = (__bf16*)alloc((size_t)H_ * H_ * 2);
  __bf16* fc_m_Wt    = (__bf16*)alloc((size_t)H_ * H_ * 2);
  __bf16* att_fc_Wt  = (__bf16*)alloc((size_t)1024 * H_ * 2);
  __bf16* att_hh_Wt  = (__bf16*)alloc((size_t)H_ * H_ * 2);
  __bf16* att_s_Wt   = (__bf16*)alloc((size_t)H_ * H_ * 2);
  __bf16* att_v_Wt   = (__bf16*)alloc((size_t)H_ * H_ * 2);
  __bf16* fc_p_Wt    = (__bf16*)alloc((size_t)V_ * H_ * 2);
  __bf16* image_bf   = (__bf16*)alloc((size_t)B_ * S_ * H_ * 2);
  float*  v_proj     = (float*) alloc((size_t)B_ * S_ * H_ * 4);
  float*  img_mean   = (float*) alloc((size_t)B_ * H_ * 4);
  __bf16* img_mean_bf= (__bf16*)alloc((size_t)B_ * H_ * 2);
  float*  h_f32      = (float*) alloc((size_t)B_ * H_ * 4);
  float*  m_f32      = (float*) alloc((size_t)B_ * H_ * 4);
  __bf16* h_bf       = (__bf16*)alloc((size_t)B_ * H_ * 2);
  float*  gates      = (float*) alloc((size_t)B_ * G4H * 4);
  float*  hs         = (float*) alloc((size_t)B_ * 1024 * 4);
  __bf16* hs_bf      = (__bf16*)alloc((size_t)B_ * 1024 * 2);
  float*  _h         = (float*) alloc((size_t)B_ * H_ * 4);
  float*  _s         = (float*) alloc((size_t)B_ * H_ * 4);
  float*  zz         = (float*) alloc((size_t)B_ * (S_ + 1) * 4);
  float*  a_buf      = (float*) alloc((size_t)B_ * (S_ + 1) * 4);
  __bf16* cp_bf      = (__bf16*)alloc((size_t)B_ * H_ * 2);
  float*  logits     = (float*) alloc((size_t)B_ * V_ * 4);

  // ---- one-time setup ----
  mean_kernel<<<B_, 256, 0, stream>>>(image, img_mean, img_mean_bf);
  static_init_kernel<<<(B_ * 1056 + 255) / 256, 256, 0, stream>>>(x_full, img_mean, view_pos, label, topic);
  wcat_kernel<<<(G4H * KCATP + 255) / 256, 256, 0, stream>>>(W_ih, W_hh, Wcat_t);
  biascat_kernel<<<(G4H + 255) / 256, 256, 0, stream>>>(b_ih, b_hh, bias_cat);
  wconv_kernel<<<(H_ * H_ + 255) / 256, 256, 0, stream>>>(fc_h_W, fc_h_Wt, H_, H_, H_);
  wconv_kernel<<<(H_ * H_ + 255) / 256, 256, 0, stream>>>(fc_m_W, fc_m_Wt, H_, H_, H_);
  wconv_kernel<<<(1024 * H_ + 255) / 256, 256, 0, stream>>>(att_fc_W, att_fc_Wt, H_, 1024, H_);
  wconv_kernel<<<(H_ * H_ + 255) / 256, 256, 0, stream>>>(att_hh_W, att_hh_Wt, H_, H_, H_);
  wconv_kernel<<<(H_ * H_ + 255) / 256, 256, 0, stream>>>(att_s_W, att_s_Wt, H_, H_, H_);
  wconv_kernel<<<(H_ * H_ + 255) / 256, 256, 0, stream>>>(att_v_W, att_v_Wt, H_, H_, H_);
  wconv_kernel<<<(V_ * H_ + 255) / 256, 256, 0, stream>>>(fc_p_W, fc_p_Wt, H_, V_, H_);
  f2bf_kernel<<<(B_ * S_ * H_ + 255) / 256, 256, 0, stream>>>(image, image_bf, B_ * S_ * H_);

  // h0 = tanh(img_mean @ fc_h_W + b);  m0 = tanh(img_mean @ fc_m_W + b)
  gemm_launch(stream, img_mean_bf, fc_h_Wt, fc_h_b, h_f32, h_bf, B_, H_, H_, H_, 0, H_, 1);
  gemm_launch(stream, img_mean_bf, fc_m_Wt, fc_m_b, m_f32, nullptr, B_, H_, H_, H_, 0, H_, 1);

  // v_proj = image @ att_v_W + b   (12544 x 512 x 512)
  gemm_launch(stream, image_bf, att_v_Wt, att_v_b, v_proj, nullptr, B_ * S_, H_, H_, H_, 0, H_, 0);

  // ---- sequential decode: 31 steps ----
  for (int t = 0; t < T_; ++t) {
    x_update_kernel<<<(B_ * 1024 + 255) / 256, 256, 0, stream>>>(x_full, emb_W, text, h_bf, t);
    // gates = [static,x_t,h] @ [W_ih;W_hh] + (b_ih+b_hh)   (64 x 2080 x 2048)
    gemm_launch(stream, x_full, Wcat_t, bias_cat, gates, nullptr, B_, G4H, KCATP, KCATP, 0, G4H, 0);
    lstm_kernel<<<(B_ * H_ + 255) / 256, 256, 0, stream>>>(gates, h_f32, m_f32, h_bf);
    // hs = tanh(h2 @ att_fc_W + b)   (64 x 512 x 1024)
    gemm_launch(stream, h_bf, att_fc_Wt, att_fc_b, hs, hs_bf, B_, 1024, H_, H_, 0, 1024, 1);
    // _h = hs[:, :512] @ att_hh_W + b ; _s = hs[:, 512:] @ att_s_W + b
    gemm_launch(stream, hs_bf, att_hh_Wt, att_hh_b, _h, nullptr, B_, H_, H_, 1024, 0,   H_, 0);
    gemm_launch(stream, hs_bf, att_s_Wt,  att_s_b,  _s, nullptr, B_, H_, H_, 1024, 512, H_, 0);
    zz_kernel<<<B_ * 25, 256, 0, stream>>>(v_proj, _h, _s, att_z_W, att_z_b, zz);
    softmax_att_kernel<<<B_, 256, 0, stream>>>(zz, a_buf, att_out, length, t);
    ctx_kernel<<<B_, 256, 0, stream>>>(a_buf, image, hs, h_f32, cp_bf);
    // logits = (c + h2) @ fc_p_W + b   (64 x 512 x 10000)
    gemm_launch(stream, cp_bf, fc_p_Wt, fc_p_b, logits, nullptr, B_, V_, H_, H_, 0, V_, 0);
    logsoftmax_kernel<<<B_, 256, 0, stream>>>(logits, logp_out, temp, length, t);
  }
}
